// EnsembleDShockSolver_343597384511
// MI455X (gfx1250) — matrix operand order repaired and checked by reading, not compiled, not run on hardware
//
#include <hip/hip_runtime.h>
#include <hip/hip_bf16.h>

typedef _Float16 half_t;
typedef __attribute__((ext_vector_type(16))) _Float16 v16h;
typedef __attribute__((ext_vector_type(8)))  _Float16 v8h;
typedef __attribute__((ext_vector_type(4)))  _Float16 v4h;
typedef __attribute__((ext_vector_type(8)))  float    v8f;

#define GAMMA_F 1.6666666666666667f

// ---- dynamic LDS layout (bytes) ----
// w2h : 8*64*64 half = 65536
// w1h : 8*6*64  half =  6144
// w3h : 8*64    half =  1024
// xh  : 64*32   half =  4096
// hbuf: 8*16*64 half = 16384
// b1f : 512 f32      =  2048
// b2f : 512 f32      =  2048
// b3f : 8   f32      =    32
// xif : 64*8 f32     =  2048
// praw: 64*6 f32     =  1536
#define OFF_W2H   0
#define OFF_W1H   65536
#define OFF_W3H   71680
#define OFF_XH    72704
#define OFF_HBUF  76800
#define OFF_B1F   93184
#define OFF_B2F   95232
#define OFF_B3F   97280
#define OFF_XIF   97312
#define OFF_PRAW  99360
#define SMEM_BYTES 100896

// ---- fast math primitives (native gfx1250 trans ops) ----
static __device__ inline float rcp_fast(float x)  { return __builtin_amdgcn_rcpf(x); }
static __device__ inline float rsq_fast(float x)  { return __builtin_amdgcn_rsqf(x); }
static __device__ inline float sqrt_fast(float x) { return __builtin_amdgcn_sqrtf(x); }
static __device__ inline float fdiv_fast(float a, float b) { return a * __builtin_amdgcn_rcpf(b); }

static __device__ inline float tanh_fast(float x) {
#if __has_builtin(__builtin_amdgcn_tanhf)
    return __builtin_amdgcn_tanhf(x);          // native V_TANH_F32 on gfx1250
#else
    x = fminf(fmaxf(x, -15.f), 15.f);
    float e = __expf(2.f * x);
    return (e - 1.f) * rcp_fast(e + 1.f);
#endif
}

static __device__ inline float sigmoid_fast(float x) {
    return rcp_fast(1.f + __expf(-x));
}

// A-fragment (16-bit, 16x32) per ISA layout:
//   lane l: M = l&15, g = l>>4
//   halves 0..7  -> K = kbase + 8*g + 0..7    (contiguous 16B)
//   halves 8..15 -> K = kbase + 16 + 8*g + 0..7 (contiguous 16B)
static __device__ inline v16h load_a_frag(const half_t* mat, int stride, int Ml, int g, int kbase) {
    const half_t* r = mat + Ml * stride + kbase + 8 * g;
    v8h lo = *(const v8h*)r;
    v8h hi = *(const v8h*)(r + 16);
    return __builtin_shufflevector(lo, hi, 0,1,2,3,4,5,6,7,8,9,10,11,12,13,14,15);
}

// bias + tanh + f16 store of a 16x64 C-tile set (C layout: lane l holds
// (M = r + 8*(l>>4), N = l&15) in v8f element r) into LDS row-major [16][64]
static __device__ inline void store_act(half_t* hs, const v8f* acc, const float* bias, int Ml, int g) {
#pragma unroll
    for (int t = 0; t < 4; ++t) {
        float bb = bias[16 * t + Ml];
#pragma unroll
        for (int r = 0; r < 8; ++r) {
            float v = acc[t][r] + bb;
            hs[(r + 8 * g) * 64 + 16 * t + Ml] = (half_t)tanh_fast(v);
        }
    }
}

// Relativistic shock jump conditions (Taub adiabat) — mirrors _get_vel_shock
static __device__ inline void shock_branch(float sign, float rho, float p, float v, float pC,
                                           float& rhoC, float& hC, float& vstar, float& vshock) {
    const float g = GAMMA_F, gm1 = g - 1.f;
    float rcp_rho = rcp_fast(rho);
    float rcp_gpC = rcp_fast(g * pC);
    float h  = 1.f + g / gm1 * p * rcp_rho;
    float W  = rsq_fast(1.f - v * v);
    float dp = p - pC;
    float A  = 1.f + gm1 * dp * rcp_gpC;
    float B  = -gm1 * dp * rcp_gpC;
    float C  = h * dp * rcp_rho - h * h;
    float disc = fmaxf(B * B - 4.f * A * C, 0.f);
    float hc = (-B + sqrt_fast(disc)) * 0.5f * rcp_fast(A);
    float rc = g * pC * rcp_fast(gm1 * (hc - 1.f));
    float j2 = fmaxf(fdiv_fast(pC - p, h * rcp_rho - hc * rcp_fast(rc)), 1e-12f);
    float jabs = sqrt_fast(j2);
    float a2 = (rho * W) * (rho * W);
    float Vs = fdiv_fast(a2 * v + sign * jabs * sqrt_fast(j2 + a2 * (1.f - v * v)), a2 + j2);
    float Ws = rsq_fast(fmaxf(1.f - Vs * Vs, 1e-12f));
    float js = sign * jabs;
    float rcp_js = rcp_fast(js);
    float vst = fdiv_fast(h * W * v + Ws * (pC - p) * rcp_js,
                          h * W + (pC - p) * (Ws * v * rcp_js + rcp_fast(rho * W)));
    rhoC = rc; hC = hc; vstar = vst; vshock = Vs;
}

#define WMMA_F16(a, b, c) \
    __builtin_amdgcn_wmma_f32_16x16x32_f16(false, (a), false, (b), (short)0, (c), false, false)

__global__ __launch_bounds__(256, 1) void shock_ensemble_kernel(
    const float* __restrict__ P,  const float* __restrict__ F,
    const float* __restrict__ W1, const float* __restrict__ B1,
    const float* __restrict__ W2, const float* __restrict__ B2,
    const float* __restrict__ W3, const float* __restrict__ B3,
    float* __restrict__ out, int N)
{
    extern __shared__ char smem[];
    half_t* w2h  = (half_t*)(smem + OFF_W2H);
    half_t* w1h  = (half_t*)(smem + OFF_W1H);
    half_t* w3h  = (half_t*)(smem + OFF_W3H);
    half_t* xh   = (half_t*)(smem + OFF_XH);
    half_t* hbuf = (half_t*)(smem + OFF_HBUF);
    float*  b1f  = (float*)(smem + OFF_B1F);
    float*  b2f  = (float*)(smem + OFF_B2F);
    float*  b3f  = (float*)(smem + OFF_B3F);
    float*  xif  = (float*)(smem + OFF_XIF);
    float*  praw = (float*)(smem + OFF_PRAW);

    const int tid  = threadIdx.x;
    const int lane = tid & 31;
    const int wv   = tid >> 5;
    const long base = (long)blockIdx.x * 64;

    // ---- stage weights to LDS as f16 (vectorized f32x4 -> f16x4) ----
    {
        const float4* s = (const float4*)W2;           // 8*64*64/4 = 8192
        for (int i = tid; i < 8192; i += 256) {
            float4 v = s[i];
            v4h h = { (half_t)v.x, (half_t)v.y, (half_t)v.z, (half_t)v.w };
            *(v4h*)(w2h + 4 * i) = h;
        }
    }
    {
        const float4* s = (const float4*)W1;           // 8*6*64/4 = 768
        for (int i = tid; i < 768; i += 256) {
            float4 v = s[i];
            v4h h = { (half_t)v.x, (half_t)v.y, (half_t)v.z, (half_t)v.w };
            *(v4h*)(w1h + 4 * i) = h;
        }
    }
    {
        const float4* s = (const float4*)W3;           // 8*64/4 = 128
        for (int i = tid; i < 128; i += 256) {
            float4 v = s[i];
            v4h h = { (half_t)v.x, (half_t)v.y, (half_t)v.z, (half_t)v.w };
            *(v4h*)(w3h + 4 * i) = h;
        }
    }
    for (int i = tid; i < 512; i += 256) { b1f[i] = B1[i]; b2f[i] = B2[i]; }
    if (tid < 8) b3f[tid] = B3[tid];

    // ---- stage per-cell features: [log rhoL, log rhoR, log pL, log pR, vL, vR] ----
    if (tid < 64) {
        long gc = base + tid;
        float rL = 1.f, rR = 1.f, pL = 1.f, pR = 1.f, vL = 0.f, vR = 0.f;
        if (gc < (long)N) {
            const float* pp = P + gc * 6;
            rL = pp[0]; rR = pp[1]; pL = pp[2]; pR = pp[3]; vL = pp[4]; vR = pp[5];
        }
        float* pr = praw + tid * 6;
        pr[0] = rL; pr[1] = rR; pr[2] = pL; pr[3] = pR; pr[4] = vL; pr[5] = vR;
        half_t* xr = xh + tid * 32;
        xr[0] = (half_t)__logf(rL); xr[1] = (half_t)__logf(rR);
        xr[2] = (half_t)__logf(pL); xr[3] = (half_t)__logf(pR);
        xr[4] = (half_t)vL;         xr[5] = (half_t)vR;
#pragma unroll
        for (int k = 6; k < 32; ++k) xr[k] = (half_t)0.f;
    }
    __syncthreads();

    // ---- per-wave MLP: wave wv computes ensemble member e = wv for 64 cells ----
    {
        const int e  = wv;
        const int Ml = lane & 15;
        const int g  = lane >> 4;
        half_t* hslice = hbuf + wv * (16 * 64);

        // Resident B fragments (B layout: N = lane&15, halves = K in A-fragment order 8g+s / 16+8g+s-8)
        v16h w1b[4], w2b[2][4], w3b[2];
#pragma unroll
        for (int t = 0; t < 4; ++t) {
            v16h b = {};
#pragma unroll
            for (int s = 0; s < 16; ++s) {
                int kk = 8 * g + s + (s & 8);
                if (kk < 6) b[s] = w1h[(e * 6 + kk) * 64 + 16 * t + Ml];
            }
            w1b[t] = b;
        }
#pragma unroll
        for (int k = 0; k < 2; ++k)
#pragma unroll
            for (int t = 0; t < 4; ++t)
#pragma unroll
                for (int s = 0; s < 16; ++s)
                    w2b[k][t][s] = w2h[(e * 64 + 32 * k + 8 * g + s + (s & 8)) * 64 + 16 * t + Ml];
#pragma unroll
        for (int k = 0; k < 2; ++k) {
            v16h b = {};
            if (Ml == 0)   // only output column N=0 is real
#pragma unroll
                for (int s = 0; s < 16; ++s)
                    b[s] = w3h[e * 64 + 32 * k + 8 * g + s + (s & 8)];
            w3b[k] = b;
        }

#pragma unroll 1
        for (int mt = 0; mt < 4; ++mt) {
            const int cbase = 16 * mt;

            // layer 1: x(16x6, K padded to 32) @ W1(6x64)
            v16h ax = load_a_frag(xh + cbase * 32, 32, Ml, g, 0);
            v8f acc1[4];
#pragma unroll
            for (int t = 0; t < 4; ++t) {
                v8f z = {};
                acc1[t] = WMMA_F16(ax, w1b[t], z);
            }
            store_act(hslice, acc1, b1f + e * 64, Ml, g);

            // layer 2: h1(16x64) @ W2(64x64)
            v16h a0 = load_a_frag(hslice, 64, Ml, g, 0);
            v16h a1 = load_a_frag(hslice, 64, Ml, g, 32);
            v8f acc2[4];
#pragma unroll
            for (int t = 0; t < 4; ++t) {
                v8f z = {};
                z = WMMA_F16(a0, w2b[0][t], z);
                acc2[t] = WMMA_F16(a1, w2b[1][t], z);
            }
            store_act(hslice, acc2, b2f + e * 64, Ml, g);

            // layer 3: h2(16x64) @ W3(64x1)  (N padded to 16; column 0 valid)
            a0 = load_a_frag(hslice, 64, Ml, g, 0);
            a1 = load_a_frag(hslice, 64, Ml, g, 32);
            v8f z3 = {};
            z3 = WMMA_F16(a0, w3b[0], z3);
            z3 = WMMA_F16(a1, w3b[1], z3);

            if (Ml == 0) {            // lanes 0 and 16 hold column N=0
                float bb = b3f[e];
#pragma unroll
                for (int r = 0; r < 8; ++r) {
                    float xi = sigmoid_fast(z3[r] + bb);
                    xif[(cbase + r + 8 * g) * 8 + e] = xi;
                }
            }
        }
    }
    __syncthreads();

    // ---- per-cell: ensemble argmin + shock selection + flux assembly ----
    if (tid < 64) {
        long gc = base + tid;
        if (gc < (long)N) {
            const float* pr = praw + tid * 6;
            float rL = pr[0], rR = pr[1], pL = pr[2], pR = pr[3], vL = pr[4], vR = pr[5];
            float amax = fmaxf(pL, pR);

            float best = 3.402823e38f;
            float b_pC = 0.f, b_rhoCL = 0.f, b_hCL = 0.f, b_vstL = 0.f, b_vshL = 0.f;
            float b_rhoCR = 0.f, b_hCR = 0.f, b_vstR = 0.f, b_vshR = 0.f;
#pragma unroll 1
            for (int e = 0; e < 8; ++e) {
                float xi = xif[tid * 8 + e];
                float pC = amax * (1.f + xi) * rcp_fast(1.f - xi);
                float rhoCL, hCL, vstL, vshL, rhoCR, hCR, vstR, vshR;
                shock_branch(-1.f, rL, pL, vL, pC, rhoCL, hCL, vstL, vshL);
                shock_branch(+1.f, rR, pR, vR, pC, rhoCR, hCR, vstR, vshR);
                float dv = fabsf(vstL - vstR);
                if (dv < best) {       // strict < : keeps first minimum (argmin semantics)
                    best = dv;
                    b_pC = pC;
                    b_rhoCL = rhoCL; b_hCL = hCL; b_vstL = vstL; b_vshL = vshL;
                    b_rhoCR = rhoCR; b_hCR = hCR; b_vstR = vstR; b_vshR = vshR;
                }
            }

            float lambdaC = 0.5f * (b_vstR + b_vstL);
            float WC = rsq_fast(1.f - lambdaC * lambdaC);
            float densCL = WC * b_rhoCL, densCR = WC * b_rhoCR;
            float momCL = WC * WC * b_rhoCL * b_hCL * lambdaC;
            float momCR = WC * WC * b_rhoCR * b_hCR * lambdaC;
            float FCL0 = densCL * lambdaC;
            float FCL1 = densCL * (WC * b_hCL - 1.f) * lambdaC;
            float FCL2 = momCL * lambdaC + b_pC;
            float FCR0 = densCR * lambdaC;
            float FCR1 = densCR * (WC * b_hCR - 1.f) * lambdaC;
            float FCR2 = momCR * lambdaC + b_pC;

            const float* Fp = F + gc * 6;   // (N,3,2): side0 at even idx, side1 odd
            float f0 = 0.f, f1 = 0.f, f2 = 0.f;
            if (b_vshL >= 0.f)                   { f0 = Fp[0]; f1 = Fp[2]; f2 = Fp[4]; }
            if (b_vshL < 0.f && lambdaC > 0.f)   { f0 = FCL0;  f1 = FCL1;  f2 = FCL2;  }
            if (lambdaC <= 0.f && b_vshR > 0.f)  { f0 = FCR0;  f1 = FCR1;  f2 = FCR2;  }
            if (b_vshR <= 0.f)                   { f0 = Fp[1]; f1 = Fp[3]; f2 = Fp[5]; }

            out[gc * 3 + 0] = f0;
            out[gc * 3 + 1] = f1;
            out[gc * 3 + 2] = f2;
        }
    }
}

extern "C" void kernel_launch(void* const* d_in, const int* in_sizes, int n_in,
                              void* d_out, int out_size, void* d_ws, size_t ws_size,
                              hipStream_t stream) {
    // setup_inputs order: P, U, F, cmax, cmin, W1, b1, W2, b2, W3, b3
    const float* P  = (const float*)d_in[0];
    const float* F  = (const float*)d_in[2];
    const float* W1 = (const float*)d_in[5];
    const float* B1 = (const float*)d_in[6];
    const float* W2 = (const float*)d_in[7];
    const float* B2 = (const float*)d_in[8];
    const float* W3 = (const float*)d_in[9];
    const float* B3 = (const float*)d_in[10];
    float* out = (float*)d_out;

    const int N = in_sizes[0] / 6;           // P is (N,3,2)
    const int nBlocks = (N + 63) / 64;

    shock_ensemble_kernel<<<nBlocks, 256, SMEM_BYTES, stream>>>(
        P, F, W1, B1, W2, B2, W3, B3, out, N);
}
